// LSTMAutoencoder_11424613007840
// MI455X (gfx1250) — compile-verified, matrix operand order checked by reading, not run on hardware
//
#include <hip/hip_runtime.h>
#include <hip/hip_bf16.h>
#include <cstdint>

// ---------------------------------------------------------------------------
// LSTM autoencoder forward for MI455X (gfx1250, wave32, WMMA 16x16x32 bf16).
//   H=256, I=8, B=64, S=2048.
// Heavy GEMMs (enc1/dec1 input projections, ~550 GFLOP) -> LDS-staged tiled
// WMMA kernel (block tile 64x256, wave tile 64x32, A slab shared via LDS).
// Sequential recurrences -> persistent 32-wave workgroup per direction using
// 288KB of the WGP's 320KB LDS (256KB f32 pre-activations + 32KB bf16 h),
// cell state in registers, WMMA for h @ U^T each step.
// ---------------------------------------------------------------------------

#define HID   256
#define BATCH 64
#define SEQ   2048
#define G4H   1024                    // 4*H
#define MROWS (BATCH * SEQ)           // 131072 flattened (b,s) rows

typedef __attribute__((ext_vector_type(16))) __bf16 v16bf;
typedef __attribute__((ext_vector_type(8)))  float  v8f;

struct U32x8 { uint4 lo, hi; };

__device__ __forceinline__ v16bf make_frag(uint4 lo, uint4 hi) {
  U32x8 t{lo, hi};
  return __builtin_bit_cast(v16bf, t);
}

__device__ __forceinline__ unsigned short f2bf(float f) {
  unsigned u = __float_as_uint(f);
  u += 0x7FFFu + ((u >> 16) & 1u);            // round-to-nearest-even
  return (unsigned short)(u >> 16);
}
__device__ __forceinline__ float bf2f(unsigned short s) {
  return __uint_as_float(((unsigned)s) << 16);
}
__device__ __forceinline__ float sigmoidf(float x) {
  return 1.0f / (1.0f + __expf(-x));
}

// ---------------------------------------------------------------------------
// Pack a row-major fp32 weight [N, K] (row = output unit, col = input) into
// WMMA B-operand bf16 fragments. Tile (kt, nt) covers K=kt*32..+32, N=nt*16..+16.
// Layout: dst[((kt*(N/16)+nt)*32 + lane)*16 + slot], mirroring the 16-bit
// A-fragment K-split (lanes 0-15: K 0..7,16..23; lanes 16-31: K 8..15,24..31).
// ---------------------------------------------------------------------------
__global__ void pack_wmma_b(const float* __restrict__ src,
                            unsigned short* __restrict__ dst, int N, int K) {
  int gid  = blockIdx.x * blockDim.x + threadIdx.x;
  int lane = gid & 31;
  int tile = gid >> 5;
  int nt16 = N >> 4;
  int ntiles = (K >> 5) * nt16;
  if (tile >= ntiles) return;
  int kt = tile / nt16;
  int nt = tile - kt * nt16;
  int n  = nt * 16 + (lane & 15);
  int kbase = kt * 32 + ((lane >> 4) ? 8 : 0);
  unsigned short* out = dst + ((size_t)tile * 32 + lane) * 16;
#pragma unroll
  for (int s = 0; s < 16; ++s) {
    int k = (s < 8) ? (kbase + s) : (kbase + s + 8);
    out[s] = f2bf(src[(size_t)n * K + k]);
  }
}

// ---------------------------------------------------------------------------
// LDS-staged tiled WMMA GEMM: C[M,N] = A[M,K](bf16) * Bpacked + bias, fp32 acc.
// 256 threads = 8 waves. Block tile 64(M) x 256(N); wave: 4 M-tiles x 2 N-tiles.
// Per K-step: the 64x32 bf16 A slab (4KB) is loaded to LDS once (coalesced,
// one b128 per thread) and A fragments come from ds_load_b128 -- removing the
// 8x redundant global A reads, global-load:wmma issue ratio = 3:8.
// ---------------------------------------------------------------------------
__global__ __launch_bounds__(256, 1)
void gemm_gx_bf16(const unsigned short* __restrict__ A,
                  const unsigned short* __restrict__ Bp,
                  const float* __restrict__ bias,
                  float* __restrict__ C,
                  int M, int N, int K) {
  __shared__ unsigned short aSm[64 * 32];      // [64 rows][32 k] bf16, 4KB

  const int tid   = threadIdx.x;
  const int lane  = tid & 31;
  const int w     = tid >> 5;                  // wave 0..7
  const int mBase = blockIdx.x * 64;
  const int nt0   = blockIdx.y * 16 + w * 2;   // two 16-col tiles per wave
  const int nt16  = N >> 4;
  const int ml    = lane & 15;
  const int mh    = lane >> 4;

  const int cpRow = tid >> 2;                  // 0..63
  const int cpCol = (tid & 3) * 8;             // 16B chunk within 32-elem row

  const v8f vzero = {0.f, 0.f, 0.f, 0.f, 0.f, 0.f, 0.f, 0.f};
  v8f acc[8];
#pragma unroll
  for (int q = 0; q < 8; ++q) acc[q] = vzero;

  const int KT = K >> 5;
  for (int kt = 0; kt < KT; ++kt) {
    __syncthreads();                           // previous slab fully consumed
    *(uint4*)(aSm + cpRow * 32 + cpCol) =
        *(const uint4*)(A + (size_t)(mBase + cpRow) * K + kt * 32 + cpCol);

    const unsigned short* bp0 = Bp + (((size_t)kt * nt16 + nt0) * 32 + lane) * 16;
    if (kt + 1 < KT)
      __builtin_prefetch(bp0 + (size_t)nt16 * 32 * 16, 0, 1);   // next K-slab
    v16bf bf0 = make_frag(*(const uint4*)bp0, *(const uint4*)(bp0 + 8));
    const unsigned short* bp1 = bp0 + 32 * 16;
    v16bf bf1 = make_frag(*(const uint4*)bp1, *(const uint4*)(bp1 + 8));
    __syncthreads();                           // slab visible

#pragma unroll
    for (int mt = 0; mt < 4; ++mt) {
      const unsigned short* ap = aSm + (mt * 16 + ml) * 32 + (mh ? 8 : 0);
      v16bf af = make_frag(*(const uint4*)ap, *(const uint4*)(ap + 16));
      acc[mt * 2 + 0] = __builtin_amdgcn_wmma_f32_16x16x32_bf16(
          false, af, false, bf0, (short)0, acc[mt * 2 + 0], false, false);
      acc[mt * 2 + 1] = __builtin_amdgcn_wmma_f32_16x16x32_bf16(
          false, af, false, bf1, (short)0, acc[mt * 2 + 1], false, false);
    }
  }

#pragma unroll
  for (int c = 0; c < 2; ++c) {
    const int n  = (nt0 + c) * 16 + ml;
    const float bv = bias ? bias[n] : 0.0f;
#pragma unroll
    for (int mt = 0; mt < 4; ++mt) {
#pragma unroll
      for (int r = 0; r < 8; ++r) {
        int m = mBase + mt * 16 + r + (mh << 3);
        C[(size_t)m * N + n] = acc[mt * 2 + c][r] + bv;
      }
    }
  }
}

// ---------------------------------------------------------------------------
// Persistent bidirectional LSTM recurrence. grid = 2 blocks (dir 0 fwd / 1 bwd),
// 1024 threads = 32 waves. Per step: g = gx[t] + h @ U^T via WMMA (K=256),
// pre-activations staged in LDS fp32, gates + cell update elementwise with c
// in registers, new h written to LDS as bf16 (A-fragments read from LDS).
//   dyn LDS = 64*1024*4 (gSm) + 64*256*2 (hSm) = 294912 B  (<= 320KB WGP LDS)
// ---------------------------------------------------------------------------
__global__ __launch_bounds__(1024, 1)
void lstm_recurrence(const float* __restrict__ gxA, const float* __restrict__ gxB,
                     const unsigned short* __restrict__ upA,
                     const unsigned short* __restrict__ upB,
                     unsigned short* __restrict__ yCat,  // [B,S,2H] bf16 or null
                     float* __restrict__ hFinal,         // [B,2H] f32 or null
                     int constGx) {
  extern __shared__ char smem[];
  float* gSm = (float*)smem;                                       // [64][1024]
  unsigned short* hSm = (unsigned short*)(smem + 64 * G4H * 4);    // [64][256]

  const int dir = blockIdx.x;
  const float* gx = dir ? gxB : gxA;
  const unsigned short* up = dir ? upB : upA;

  const int tid  = threadIdx.x;
  const int lane = tid & 31;
  const int w    = tid >> 5;       // wave 0..31
  const int ml   = lane & 15;
  const int mh   = lane >> 4;
  const int nt0  = w * 2;          // each wave owns two 16-col tiles of 4H

  // elementwise ownership: thread -> (j = hidden unit, 16 batch rows)
  const int j    = tid & 255;
  const int bgrp = tid >> 8;       // 0..3
  float creg[16];
#pragma unroll
  for (int i = 0; i < 16; ++i) creg[i] = 0.0f;

  for (int idx = tid; idx < BATCH * HID; idx += 1024) hSm[idx] = 0;

  const v8f vzero = {0.f, 0.f, 0.f, 0.f, 0.f, 0.f, 0.f, 0.f};

  for (int t = 0; t < SEQ; ++t) {
    const int tt = dir ? (SEQ - 1 - t) : t;
    __syncthreads();               // h(t-1) visible in hSm

    v8f acc[8];
#pragma unroll
    for (int q = 0; q < 8; ++q) acc[q] = vzero;

#pragma unroll
    for (int kt = 0; kt < 8; ++kt) {
      const unsigned short* bp0 = up + (((size_t)kt * 64 + nt0) * 32 + lane) * 16;
      if (kt < 7)
        __builtin_prefetch(bp0 + (size_t)64 * 32 * 16, 0, 2);  // next K-slab (L2-hot)
      v16bf bf0 = make_frag(*(const uint4*)bp0, *(const uint4*)(bp0 + 8));
      const unsigned short* bp1 = bp0 + 32 * 16;
      v16bf bf1 = make_frag(*(const uint4*)bp1, *(const uint4*)(bp1 + 8));
#pragma unroll
      for (int mt = 0; mt < 4; ++mt) {
        const unsigned short* ap =
            hSm + (mt * 16 + ml) * HID + kt * 32 + (mh ? 8 : 0);
        v16bf af = make_frag(*(const uint4*)ap, *(const uint4*)(ap + 16));
        acc[mt * 2 + 0] = __builtin_amdgcn_wmma_f32_16x16x32_bf16(
            false, af, false, bf0, (short)0, acc[mt * 2 + 0], false, false);
        acc[mt * 2 + 1] = __builtin_amdgcn_wmma_f32_16x16x32_bf16(
            false, af, false, bf1, (short)0, acc[mt * 2 + 1], false, false);
      }
    }

    // stage pre-activations h@U^T into LDS
#pragma unroll
    for (int mt = 0; mt < 4; ++mt) {
#pragma unroll
      for (int c = 0; c < 2; ++c) {
        int n = (nt0 + c) * 16 + ml;
#pragma unroll
        for (int r = 0; r < 8; ++r) {
          int brow = mt * 16 + r + (mh << 3);
          gSm[brow * G4H + n] = acc[mt * 2 + c][r];
        }
      }
    }
    __syncthreads();

    // gates: i,f,g,o at j, 256+j, 512+j, 768+j
#pragma unroll
    for (int i = 0; i < 16; ++i) {
      const int b = bgrp * 16 + i;
      float gi = gSm[b * G4H + j];
      float gf = gSm[b * G4H + HID + j];
      float gg = gSm[b * G4H + 2 * HID + j];
      float go = gSm[b * G4H + 3 * HID + j];
      const size_t base =
          constGx ? (size_t)b * G4H : ((size_t)b * SEQ + tt) * G4H;
      gi += gx[base + j];
      gf += gx[base + HID + j];
      gg += gx[base + 2 * HID + j];
      go += gx[base + 3 * HID + j];
      float cN = sigmoidf(gf) * creg[i] + sigmoidf(gi) * tanhf(gg);
      creg[i] = cN;
      float h = sigmoidf(go) * tanhf(cN);
      hSm[b * HID + j] = f2bf(h);
      if (yCat)
        yCat[((size_t)b * SEQ + tt) * (2 * HID) + dir * HID + j] = f2bf(h);
    }
  }

  if (hFinal) {
#pragma unroll
    for (int i = 0; i < 16; ++i) {
      const int b = bgrp * 16 + i;
      hFinal[(size_t)b * (2 * HID) + dir * HID + j] = bf2f(hSm[b * HID + j]);
    }
  }
}

// ---------------------------------------------------------------------------
// Encoder layer-0 input projection: K=8 -> plain VALU (WMMA would waste 3/4).
// ---------------------------------------------------------------------------
__global__ void enc0_gx(const float* __restrict__ x, const float* __restrict__ W,
                        const float* __restrict__ b, float* __restrict__ gx) {
  size_t gid = (size_t)blockIdx.x * blockDim.x + threadIdx.x;
  if (gid >= (size_t)MROWS * G4H) return;
  int n = (int)(gid & (G4H - 1));
  size_t m = gid >> 10;
  float s = b[n];
#pragma unroll
  for (int k = 0; k < 8; ++k) s += x[m * 8 + k] * W[n * 8 + k];
  gx[gid] = s;
}

// Decoder layer-0 input projection: context is constant over S, so compute
// gx0[B,4H] = ctx @ W^T + b exactly once (saves a 137-GFLOP GEMM).
__global__ void ctx_gx(const float* __restrict__ ctx, const float* __restrict__ W,
                       const float* __restrict__ b, float* __restrict__ gx0) {
  int gid = blockIdx.x * blockDim.x + threadIdx.x;
  int n = gid & (G4H - 1);
  int bb = gid >> 10;
  if (bb >= BATCH) return;
  float s = b[n];
  for (int k = 0; k < 2 * HID; ++k)
    s += ctx[bb * 2 * HID + k] * W[(size_t)n * 2 * HID + k];
  gx0[gid] = s;
}

// Output projection: N=8 -> VALU dot products.
__global__ void out_proj(const unsigned short* __restrict__ y,
                         const float* __restrict__ Wo,
                         const float* __restrict__ bo, float* __restrict__ out) {
  size_t gid = (size_t)blockIdx.x * blockDim.x + threadIdx.x;
  if (gid >= (size_t)MROWS * 8) return;
  int o = (int)(gid & 7);
  size_t m = gid >> 3;
  float s = bo[o];
  for (int k = 0; k < 2 * HID; ++k)
    s += bf2f(y[m * 2 * HID + k]) * Wo[o * 2 * HID + k];
  out[gid] = s;
}

// ---------------------------------------------------------------------------
extern "C" void kernel_launch(void* const* d_in, const int* in_sizes, int n_in,
                              void* d_out, int out_size, void* d_ws, size_t ws_size,
                              hipStream_t stream) {
  (void)in_sizes; (void)n_in; (void)out_size; (void)ws_size;
  const float* x = (const float*)d_in[0];
  // dict order: e0f, e0b, e1f, e1b, d0f, d0b, d1f, d1b  (W,U,b each), out_W, out_b
  const float* W_[8];
  const float* U_[8];
  const float* b_[8];
  for (int i = 0; i < 8; ++i) {
    W_[i] = (const float*)d_in[1 + 3 * i + 0];
    U_[i] = (const float*)d_in[1 + 3 * i + 1];
    b_[i] = (const float*)d_in[1 + 3 * i + 2];
  }
  const float* out_W = (const float*)d_in[25];
  const float* out_b = (const float*)d_in[26];

  // -------- workspace carving --------
  char* ws = (char*)d_ws;
  size_t off = 0;
  auto alloc = [&](size_t bytes) -> char* {
    char* p = ws + off;
    off += (bytes + 255) & ~(size_t)255;
    return p;
  };
  unsigned short* uP[8];
  for (int i = 0; i < 8; ++i)
    uP[i] = (unsigned short*)alloc((size_t)G4H * HID * 2);       // 512KB each
  unsigned short* wP[4];
  for (int i = 0; i < 4; ++i)
    wP[i] = (unsigned short*)alloc((size_t)G4H * 512 * 2);       // 1MB each
  float* gxF = (float*)alloc((size_t)MROWS * G4H * 4);           // 512MB
  float* gxB = (float*)alloc((size_t)MROWS * G4H * 4);           // 512MB
  unsigned short* yCat  = (unsigned short*)alloc((size_t)MROWS * 512 * 2); // 128MB
  unsigned short* yCat2 = (unsigned short*)alloc((size_t)MROWS * 512 * 2); // 128MB
  float* ctx  = (float*)alloc((size_t)BATCH * 512 * 4);
  float* gx0F = (float*)alloc((size_t)BATCH * G4H * 4);
  float* gx0B = (float*)alloc((size_t)BATCH * G4H * 4);

  // -------- one-time weight packing into WMMA-B fragment layout --------
  {
    int tilesU = (HID / 32) * (G4H / 16);           // 512 tiles
    int blkU = (tilesU * 32 + 255) / 256;
    for (int i = 0; i < 8; ++i)
      pack_wmma_b<<<blkU, 256, 0, stream>>>(U_[i], uP[i], G4H, HID);
    const float* bigW[4] = {W_[2], W_[3], W_[6], W_[7]};  // e1f,e1b,d1f,d1b
    int tilesW = (512 / 32) * (G4H / 16);           // 1024 tiles
    int blkW = (tilesW * 32 + 255) / 256;
    for (int i = 0; i < 4; ++i)
      pack_wmma_b<<<blkW, 256, 0, stream>>>(bigW[i], wP[i], G4H, 512);
  }

  const size_t ldsRec = (size_t)64 * G4H * 4 + (size_t)64 * HID * 2; // 294912 B
  dim3 ggrid(MROWS / 64, G4H / 256);   // block tile 64 x 256

  // -------- encoder layer 0 --------
  {
    size_t tot = (size_t)MROWS * G4H;
    unsigned blk = (unsigned)((tot + 255) / 256);
    enc0_gx<<<blk, 256, 0, stream>>>(x, W_[0], b_[0], gxF);
    enc0_gx<<<blk, 256, 0, stream>>>(x, W_[1], b_[1], gxB);
  }
  lstm_recurrence<<<2, 1024, ldsRec, stream>>>(gxF, gxB, uP[0], uP[1],
                                               yCat, nullptr, 0);

  // -------- encoder layer 1 (only final hiddens needed) --------
  gemm_gx_bf16<<<ggrid, 256, 0, stream>>>(yCat, wP[0], b_[2], gxF, MROWS, G4H, 512);
  gemm_gx_bf16<<<ggrid, 256, 0, stream>>>(yCat, wP[1], b_[3], gxB, MROWS, G4H, 512);
  lstm_recurrence<<<2, 1024, ldsRec, stream>>>(gxF, gxB, uP[2], uP[3],
                                               nullptr, ctx, 0);

  // -------- decoder layer 0 (constant gx from broadcast context) --------
  ctx_gx<<<(BATCH * G4H) / 256, 256, 0, stream>>>(ctx, W_[4], b_[4], gx0F);
  ctx_gx<<<(BATCH * G4H) / 256, 256, 0, stream>>>(ctx, W_[5], b_[5], gx0B);
  lstm_recurrence<<<2, 1024, ldsRec, stream>>>(gx0F, gx0B, uP[4], uP[5],
                                               yCat, nullptr, 1);

  // -------- decoder layer 1 --------
  gemm_gx_bf16<<<ggrid, 256, 0, stream>>>(yCat, wP[2], b_[6], gxF, MROWS, G4H, 512);
  gemm_gx_bf16<<<ggrid, 256, 0, stream>>>(yCat, wP[3], b_[7], gxB, MROWS, G4H, 512);
  lstm_recurrence<<<2, 1024, ldsRec, stream>>>(gxF, gxB, uP[6], uP[7],
                                               yCat2, nullptr, 0);

  // -------- output projection --------
  {
    size_t tot = (size_t)MROWS * 8;
    unsigned blk = (unsigned)((tot + 255) / 256);
    out_proj<<<blk, 256, 0, stream>>>(yCat2, out_W, out_b, (float*)d_out);
  }
}
